// SlidingWindowMHA_2508260901280
// MI455X (gfx1250) — matrix operand order repaired and checked by reading, not compiled, not run on hardware
//
#include <hip/hip_runtime.h>
#include <hip/hip_bf16.h>

typedef _Float16 v16h __attribute__((ext_vector_type(16)));
typedef _Float16 v8h  __attribute__((ext_vector_type(8)));
typedef float    v8f  __attribute__((ext_vector_type(8)));
typedef int      v4i  __attribute__((ext_vector_type(4)));

typedef __attribute__((address_space(1))) v4i as1_v4i;   // global
typedef __attribute__((address_space(3))) v4i as3_v4i;   // LDS

#define B_  4
#define T_  1024
#define C_  1024
#define H_  16
#define D_  64
#define KS_ 16
#define NZJ 80   // wei[:, j] == 0 exactly for j >= D_ + KS_ = 80

// ---------------------------------------------------------------------------
// CDNA5 async global->LDS copy (ASYNCcnt-tracked), with portable fallbacks.
// ---------------------------------------------------------------------------
__device__ __forceinline__ void async_copy_b128(void* lds_dst, const void* gsrc) {
#if __has_builtin(__builtin_amdgcn_global_load_async_to_lds_b128)
  __builtin_amdgcn_global_load_async_to_lds_b128(
      (as1_v4i*)gsrc, (as3_v4i*)lds_dst, /*offset=*/0, /*cpol=*/0);
#else
  *(v4i*)lds_dst = *(const v4i*)gsrc;   // synchronous fallback
#endif
}

__device__ __forceinline__ void wait_async() {
#if __has_builtin(__builtin_amdgcn_s_wait_asynccnt)
  __builtin_amdgcn_s_wait_asynccnt(0);
#else
  asm volatile("s_wait_asynccnt 0x0" ::: "memory");
#endif
}

// ---------------------------------------------------------------------------
// f32 -> f16 conversion (grid-stride)
// ---------------------------------------------------------------------------
__global__ void cvt_f32_f16(const float* __restrict__ src,
                            _Float16* __restrict__ dst, int n) {
  int i = blockIdx.x * blockDim.x + threadIdx.x;
  int stride = gridDim.x * blockDim.x;
  for (; i < n; i += stride) dst[i] = (_Float16)src[i];
}

// ---------------------------------------------------------------------------
// Y[M,N] = A[M,K] @ W[N,K]^T (+ bias[N]), f16 in, f32 accumulate/out.
// Block = 256 threads = 8 waves; block tile 128x64; K step 32.
// B tile (32k x 64 cols, 4KB) is shared by all 8 waves -> staged in LDS with
// async global->LDS b128 copies, double buffered; A frags loaded from global.
// Fragment layouts used here (wave32; A/B in contiguous k, not the ISA table order):
//   A 16x32 f16 : lane L(<16) row=L, K={0..15}; lane L+16 K={16..31}
//   B 32x16 f16 : lane L col=L&15, 16 consecutive K starting at (L<16?0:16)
//   D 16x16 f32 : VGPR r -> M = r + 8*(L>=16), N = L&15
// ---------------------------------------------------------------------------
__global__ __launch_bounds__(256)
void gemm_wmma_f16(const _Float16* __restrict__ A, const _Float16* __restrict__ W,
                   const float* __restrict__ bias, float* __restrict__ Y,
                   int M, int N, int K, int has_bias) {
  __shared__ _Float16 Bs[2][64][32];   // [buf][col][k] : 4KB per buffer

  const int tid  = threadIdx.x;
  const int wave = tid >> 5;
  const int lane = tid & 31;
  const int r    = lane & 15;
  const int half = lane >> 4;

  const int rowBase = blockIdx.y * 128 + wave * 16;
  const int colBase = blockIdx.x * 64;

  // async B loader mapping: thread t -> col = t/4, 16B segment = t%4
  const int lcol = tid >> 2;
  const int lseg = tid & 3;
  const _Float16* wsrc = W + (size_t)(colBase + lcol) * K + lseg * 8;
  _Float16* ldst[2] = { &Bs[0][lcol][lseg * 8], &Bs[1][lcol][lseg * 8] };

  // prologue: stage k-step 0 into buffer 0
  async_copy_b128(ldst[0], wsrc);
  wait_async();
  __syncthreads();

  v8f acc[4];
  #pragma unroll
  for (int c = 0; c < 4; ++c) acc[c] = (v8f){};

  const _Float16* arow = A + (size_t)(rowBase + r) * K + half * 16;

  const int nk = K / 32;
  for (int kt = 0; kt < nk; ++kt) {
    const int k0 = kt * 32;
    // prefetch next B tile into the other buffer (prev barrier guarantees
    // nobody is still reading it)
    if (kt + 1 < nk)
      async_copy_b128(ldst[(kt + 1) & 1], wsrc + (size_t)(kt + 1) * 32);

    // A fragment straight from global (per-wave unique rows)
    v8h alo = *(const v8h*)(arow + k0);
    v8h ahi = *(const v8h*)(arow + k0 + 8);
    v16h a;
    #pragma unroll
    for (int i = 0; i < 8; ++i) { a[i] = alo[i]; a[i + 8] = ahi[i]; }

    // preload all four B fragments so the WMMAs can issue back-to-back
    const _Float16 (*Bc)[32] = Bs[kt & 1];
    v16h bfrag[4];
    #pragma unroll
    for (int c = 0; c < 4; ++c)
      bfrag[c] = *(const v16h*)&Bc[16 * c + r][half * 16];  // 32B, aligned

    #pragma unroll
    for (int c = 0; c < 4; ++c)
      acc[c] = __builtin_amdgcn_wmma_f32_16x16x32_f16(
          false, a, false, bfrag[c], (short)0, acc[c], false, false);

    wait_async();      // next tile landed
    __syncthreads();   // all waves done reading current tile
  }

  #pragma unroll
  for (int c = 0; c < 4; ++c) {
    const int col = colBase + 16 * c + r;
    const float bv = has_bias ? bias[col] : 0.0f;
    #pragma unroll
    for (int vr = 0; vr < 8; ++vr) {
      const int row = rowBase + vr + 8 * half;
      Y[(size_t)row * N + col] = acc[c][vr] + bv;
    }
  }
}

// ---------------------------------------------------------------------------
// Vtail[b,h,d] = sum_{j=NZJ}^{T-1} vp[b, j, h*D+d]   (vp in (B,T,C) layout)
// ---------------------------------------------------------------------------
__global__ void vtail_kernel(const float* __restrict__ vp, float* __restrict__ vt) {
  const int bh = blockIdx.x;
  const int d  = threadIdx.x;
  const int b = bh / H_, h = bh % H_;
  const float* p = vp + ((size_t)b * T_) * C_ + h * D_ + d;
  float s = 0.0f;
  for (int j = NZJ; j < T_; ++j) s += p[(size_t)j * C_];
  vt[(size_t)bh * D_ + d] = s;
}

// ---------------------------------------------------------------------------
// Attention core: one wave per (b,h,i) row.
//   p[d]   = qp[d]*kp[d]/8,  cs = exclusive cumsum over d (65 entries)
//   w[j]   = cs[min(j+16,64)] - cs[max(j-16,0)]   (nonzero only j<80)
//   softmax over 1024 logits = 80 computed + 944 exact zeros
//   out[d] = sum_{j<80} a_j vp[j,d] + (e^{-m}/Z) * Vtail[d]
// All 8 waves of a block share (b,h); the 80x64 f32 V block (20KB) is staged
// into LDS with async copies issued early, waited just before consumption.
// ---------------------------------------------------------------------------
__global__ __launch_bounds__(256)
void attn_kernel(const float* __restrict__ qp, const float* __restrict__ kp,
                 const float* __restrict__ vp, const float* __restrict__ vt,
                 _Float16* __restrict__ ah) {
  __shared__ float Vs[NZJ][D_];   // 20KB shared V rows j=0..79 for this (b,h)
  __shared__ float scs[8][72];    // 65 cumsum entries per wave-row (padded)
  __shared__ float sa[8][NZJ];    // 80 attention weights per wave-row

  const int tid  = threadIdx.x;
  const int wave = tid >> 5;
  const int lane = tid & 31;
  const int rowId = blockIdx.x * 8 + wave;
  const int b = rowId >> 14;             // / (H_*T_)
  const int h = (rowId >> 10) & (H_ - 1);
  const int i = rowId & (T_ - 1);

  // ---- issue async stage of V rows (same (b,h) for whole block) ----
  const float* vsrc = vp + ((size_t)b * T_) * C_ + h * D_;
  #pragma unroll
  for (int c = tid; c < NZJ * 16; c += 256) {       // 1280 x 16B chunks
    const int j = c >> 4, seg = c & 15;
    async_copy_b128(&Vs[j][seg * 4], vsrc + (size_t)j * C_ + seg * 4);
  }

  // ---- windowed-cumsum logits (overlaps the async copy) ----
  const size_t base = ((size_t)(b * T_ + i)) * C_ + h * D_;
  const float scale = 0.125f;            // 1/sqrt(D)
  float p0 = qp[base + lane]      * kp[base + lane]      * scale;
  float p1 = qp[base + lane + 32] * kp[base + lane + 32] * scale;

  float s0 = p0, s1 = p1;
  #pragma unroll
  for (int off = 1; off < 32; off <<= 1) {
    float t0 = __shfl_up(s0, off, 32);
    float t1 = __shfl_up(s1, off, 32);
    if (lane >= off) { s0 += t0; s1 += t1; }
  }
  float tot0 = __shfl(s0, 31, 32);
  if (lane == 0) scs[wave][0] = 0.0f;
  scs[wave][lane + 1]  = s0;
  scs[wave][lane + 33] = tot0 + s1;
  __syncthreads();

  const int j0 = lane, j1 = lane + 32, j2 = lane + 64;
  float w0 = scs[wave][min(j0 + KS_, D_)] - scs[wave][max(j0 - KS_, 0)];
  float w1 = scs[wave][min(j1 + KS_, D_)] - scs[wave][max(j1 - KS_, 0)];
  float w2 = (lane < 16)
           ? scs[wave][min(j2 + KS_, D_)] - scs[wave][max(j2 - KS_, 0)]
           : -__builtin_inff();

  float mloc = fmaxf(fmaxf(w0, w1), w2);
  #pragma unroll
  for (int off = 16; off > 0; off >>= 1)
    mloc = fmaxf(mloc, __shfl_xor(mloc, off, 32));
  const float m = fmaxf(mloc, 0.0f);     // include the 944 zero logits

  float e0 = __expf(w0 - m);
  float e1 = __expf(w1 - m);
  float e2 = (lane < 16) ? __expf(w2 - m) : 0.0f;
  float sloc = e0 + e1 + e2;
  #pragma unroll
  for (int off = 16; off > 0; off >>= 1)
    sloc += __shfl_xor(sloc, off, 32);
  const float em = __expf(-m);
  const float Z  = sloc + (float)(T_ - NZJ) * em;
  const float rz = 1.0f / Z;

  sa[wave][lane]      = e0 * rz;
  sa[wave][lane + 32] = e1 * rz;
  if (lane < 16) sa[wave][lane + 64] = e2 * rz;

  wait_async();        // V rows landed (own loads) ...
  __syncthreads();     // ... and everyone's, plus sa[] visible

  const float ct = em * rz;              // shared weight of every tail column
  const float* vtp = vt + ((size_t)(b * H_ + h)) * D_;
  float o0 = ct * vtp[lane];
  float o1 = ct * vtp[lane + 32];

  #pragma unroll 4
  for (int j = 0; j < NZJ; ++j) {
    const float aj = sa[wave][j];
    o0 += aj * Vs[j][lane];
    o1 += aj * Vs[j][lane + 32];
  }
  ah[base + lane]      = (_Float16)o0;
  ah[base + lane + 32] = (_Float16)o1;
}

// ---------------------------------------------------------------------------
extern "C" void kernel_launch(void* const* d_in, const int* in_sizes, int n_in,
                              void* d_out, int out_size, void* d_ws, size_t ws_size,
                              hipStream_t stream) {
  (void)in_sizes; (void)n_in; (void)out_size; (void)ws_size;
  const float* q  = (const float*)d_in[0];
  const float* k  = (const float*)d_in[1];
  const float* v  = (const float*)d_in[2];
  const float* Wq = (const float*)d_in[3];
  const float* Wk = (const float*)d_in[4];
  const float* Wv = (const float*)d_in[5];
  const float* Wp = (const float*)d_in[6];
  const float* bp = (const float*)d_in[7];

  const int M  = B_ * T_;     // 4096
  const int nX = M * C_;      // 4M elems
  const int nW = C_ * C_;     // 1M elems

  char* ws = (char*)d_ws;
  size_t off = 0;
  auto alloc = [&](size_t bytes) -> void* {
    void* p = ws + off;
    off = (off + bytes + 255) & ~(size_t)255;
    return p;
  };
  _Float16* qh  = (_Float16*)alloc((size_t)nX * 2);
  _Float16* kh  = (_Float16*)alloc((size_t)nX * 2);
  _Float16* vh  = (_Float16*)alloc((size_t)nX * 2);
  _Float16* Wqh = (_Float16*)alloc((size_t)nW * 2);
  _Float16* Wkh = (_Float16*)alloc((size_t)nW * 2);
  _Float16* Wvh = (_Float16*)alloc((size_t)nW * 2);
  _Float16* Wph = (_Float16*)alloc((size_t)nW * 2);
  float*    qp  = (float*)alloc((size_t)nX * 4);
  float*    kp  = (float*)alloc((size_t)nX * 4);
  float*    vp  = (float*)alloc((size_t)nX * 4);
  float*    vt  = (float*)alloc((size_t)B_ * H_ * D_ * 4);
  _Float16* ah  = (_Float16*)alloc((size_t)nX * 2);

  // 1) precision conversion
  cvt_f32_f16<<<2048, 256, 0, stream>>>(q,  qh,  nX);
  cvt_f32_f16<<<2048, 256, 0, stream>>>(k,  kh,  nX);
  cvt_f32_f16<<<2048, 256, 0, stream>>>(v,  vh,  nX);
  cvt_f32_f16<<<1024, 256, 0, stream>>>(Wq, Wqh, nW);
  cvt_f32_f16<<<1024, 256, 0, stream>>>(Wk, Wkh, nW);
  cvt_f32_f16<<<1024, 256, 0, stream>>>(Wv, Wvh, nW);
  cvt_f32_f16<<<1024, 256, 0, stream>>>(Wp, Wph, nW);

  // 2) q/k/v projections (WMMA + async LDS staging)
  dim3 ggrid(C_ / 64, M / 128);
  gemm_wmma_f16<<<ggrid, 256, 0, stream>>>(qh, Wqh, nullptr, qp, M, C_, C_, 0);
  gemm_wmma_f16<<<ggrid, 256, 0, stream>>>(kh, Wkh, nullptr, kp, M, C_, C_, 0);
  gemm_wmma_f16<<<ggrid, 256, 0, stream>>>(vh, Wvh, nullptr, vp, M, C_, C_, 0);

  // 3) tail V sums
  vtail_kernel<<<B_ * H_, D_, 0, stream>>>(vp, vt);

  // 4) attention core (one wave per row)
  attn_kernel<<<(B_ * H_ * T_) / 8, 256, 0, stream>>>(qp, kp, vp, vt, ah);

  // 5) output projection + bias (WMMA)
  gemm_wmma_f16<<<ggrid, 256, 0, stream>>>(ah, Wph, bp, (float*)d_out, M, C_, C_, 1);
}
